// MemoryEfficientCrossAttention_37374805409839
// MI455X (gfx1250) — matrix operand order repaired and checked by reading, not compiled, not run on hardware
//
#include <hip/hip_runtime.h>
#include <hip/hip_bf16.h>
#include <stdint.h>

// ---------------------------------------------------------------------------
// MemoryEfficientCrossAttention on gfx1250 (MI455X), bf16 WMMA pipeline.
//   prep:   f32->bf16 converts (activations) + transposed bf16 weights
//   gemm:   pure bf16 WMMA, contiguous b128 operand loads, no inner-loop cvt
//   attn:   flash attention, one wave per (b, h, 64-query tile)
// ---------------------------------------------------------------------------

typedef __attribute__((ext_vector_type(16))) __bf16 v16bf;
typedef __attribute__((ext_vector_type(2)))  __bf16 v2bf;
typedef __attribute__((ext_vector_type(8)))  float  v8f;

#define HIDDEN   1024
#define HEADS    16
#define HEAD_DIM 64
#define BATCH    2
#define QLEN     2048
#define KLEN     4096

struct Frag { uint32_t w[8]; };   // 16 bf16 per lane = A(16x32)/B(32x16) operand

__device__ inline uint16_t f2bf(float f) {
  __bf16 h = (__bf16)f;                       // hardware cvt, RNE
  return __builtin_bit_cast(uint16_t, h);
}

__device__ inline uint32_t pk_bf16(float lo, float hi) {
#if __has_builtin(__builtin_amdgcn_cvt_pk_bf16_f32)
  v2bf v = __builtin_amdgcn_cvt_pk_bf16_f32(lo, hi);
  return __builtin_bit_cast(uint32_t, v);
#else
  return ((uint32_t)f2bf(hi) << 16) | (uint32_t)f2bf(lo);
#endif
}

__device__ inline v8f wmma_bf16(const Frag& a, const Frag& b, v8f c) {
  return __builtin_amdgcn_wmma_f32_16x16x32_bf16(
      false, __builtin_bit_cast(v16bf, a),
      false, __builtin_bit_cast(v16bf, b),
      (short)0, c, false, false);
}

__device__ inline void ld_frag(Frag& f, const uint16_t* p) {
  const uint4 lo4 = *(const uint4*)(p);
  const uint4 hi4 = *(const uint4*)(p + 16);
  f.w[0] = lo4.x; f.w[1] = lo4.y; f.w[2] = lo4.z; f.w[3] = lo4.w;
  f.w[4] = hi4.x; f.w[5] = hi4.y; f.w[6] = hi4.z; f.w[7] = hi4.w;
}

// ---------------------------------------------------------------------------
// Elementwise f32 -> bf16 (8 elements per thread, b128 in / b128 out)
// ---------------------------------------------------------------------------
__global__ __launch_bounds__(256)
void cvt_bf16_kernel(const float* __restrict__ in, uint16_t* __restrict__ out, int n8) {
  const int i = blockIdx.x * 256 + threadIdx.x;
  if (i >= n8) return;
  const float4 x0 = *(const float4*)(in + (size_t)i * 8);
  const float4 x1 = *(const float4*)(in + (size_t)i * 8 + 4);
  uint4 p;
  p.x = pk_bf16(x0.x, x0.y); p.y = pk_bf16(x0.z, x0.w);
  p.z = pk_bf16(x1.x, x1.y); p.w = pk_bf16(x1.z, x1.w);
  *(uint4*)(out + (size_t)i * 8) = p;
}

// ---------------------------------------------------------------------------
// f32 W(KxN) -> bf16 W^T(NxK).  Coalesced reads; tiny one-shot kernel.
// ---------------------------------------------------------------------------
__global__ __launch_bounds__(256)
void cvt_tr_bf16_kernel(const float* __restrict__ W, uint16_t* __restrict__ Wt,
                        int K, int N) {
  const int idx = blockIdx.x * 256 + threadIdx.x;
  if (idx >= K * N) return;
  const int k = idx / N;
  const int n = idx - k * N;
  Wt[(size_t)n * K + k] = f2bf(W[idx]);
}

// ---------------------------------------------------------------------------
// GEMM: C(MxN) = A(MxK,bf16 row-major) * B(KxN) given as Bt = B^T (NxK bf16).
// One wave32 computes a 64x64 tile; inner loop = 16 b128 loads + 16 WMMA.
//   c_mode: 0 -> f32 row-major; 1 -> bf16 row-major; 2 -> bf16 transposed
//           (mode 2: out[n*ldc + m], ldc = M; store is a contiguous b128)
// ---------------------------------------------------------------------------
__global__ __launch_bounds__(32)
void gemm_bf16_kernel(const uint16_t* __restrict__ A, int lda,
                      const uint16_t* __restrict__ Bt, int ldb,
                      void* __restrict__ Cp, int c_mode, int ldc, int Kd)
{
  const int lane = threadIdx.x & 31;
  const int lr   = lane & 15;
  const int hi   = lane >> 4;
  const int m0   = blockIdx.y * 64;
  const int n0   = blockIdx.x * 64;

  v8f acc[4][4];
#pragma unroll
  for (int i = 0; i < 4; ++i)
#pragma unroll
    for (int j = 0; j < 4; ++j)
#pragma unroll
      for (int r = 0; r < 8; ++r) acc[i][j][r] = 0.0f;

  const uint16_t* arow[4];
  const uint16_t* brow[4];
#pragma unroll
  for (int mt = 0; mt < 4; ++mt)
    arow[mt] = A + (size_t)(m0 + mt * 16 + lr) * lda + hi * 8;
#pragma unroll
  for (int nt = 0; nt < 4; ++nt)
    brow[nt] = Bt + (size_t)(n0 + nt * 16 + lr) * ldb + hi * 8;

  for (int k0 = 0; k0 < Kd; k0 += 32) {
    Frag afr[4], bfr[4];
#pragma unroll
    for (int mt = 0; mt < 4; ++mt) {
      ld_frag(afr[mt], arow[mt] + k0);
      __builtin_prefetch(arow[mt] + k0 + 256, 0, 0);   // next K panels
    }
#pragma unroll
    for (int nt = 0; nt < 4; ++nt) {
      ld_frag(bfr[nt], brow[nt] + k0);
      __builtin_prefetch(brow[nt] + k0 + 256, 0, 0);
    }
#pragma unroll
    for (int mt = 0; mt < 4; ++mt)
#pragma unroll
      for (int nt = 0; nt < 4; ++nt)
        acc[mt][nt] = wmma_bf16(afr[mt], bfr[nt], acc[mt][nt]);
  }

  // ---- store (C layout: lane holds column n = nt*16+lr, rows m = mt*16 + 8*hi + r)
  if (c_mode == 0) {
    float* Co = (float*)Cp;
#pragma unroll
    for (int mt = 0; mt < 4; ++mt)
#pragma unroll
      for (int nt = 0; nt < 4; ++nt) {
        const int n = n0 + nt * 16 + lr;
#pragma unroll
        for (int r = 0; r < 8; ++r) {
          const int m = m0 + mt * 16 + 8 * hi + r;
          Co[(size_t)m * ldc + n] = acc[mt][nt][r];
        }
      }
  } else if (c_mode == 1) {
    uint16_t* Co = (uint16_t*)Cp;
#pragma unroll
    for (int mt = 0; mt < 4; ++mt)
#pragma unroll
      for (int nt = 0; nt < 4; ++nt) {
        const int n = n0 + nt * 16 + lr;
#pragma unroll
        for (int r = 0; r < 8; ++r) {
          const int m = m0 + mt * 16 + 8 * hi + r;
          Co[(size_t)m * ldc + n] = f2bf(acc[mt][nt][r]);
        }
      }
  } else {  // transposed bf16: 8 consecutive m per lane -> single b128 store
    uint16_t* Co = (uint16_t*)Cp;
#pragma unroll
    for (int mt = 0; mt < 4; ++mt)
#pragma unroll
      for (int nt = 0; nt < 4; ++nt) {
        const int n  = n0 + nt * 16 + lr;
        const int mb = m0 + mt * 16 + 8 * hi;
        uint4 pkd;
        pkd.x = pk_bf16(acc[mt][nt][0], acc[mt][nt][1]);
        pkd.y = pk_bf16(acc[mt][nt][2], acc[mt][nt][3]);
        pkd.z = pk_bf16(acc[mt][nt][4], acc[mt][nt][5]);
        pkd.w = pk_bf16(acc[mt][nt][6], acc[mt][nt][7]);
        *(uint4*)(Co + (size_t)n * ldc + mb) = pkd;
      }
  }
}

// ---------------------------------------------------------------------------
// Flash attention: one wave32 per (b, h, 64-query tile), streaming 64-key
// chunks with online softmax.  P (f32 C-layout) -> LDS bf16 -> A-layout.
// ---------------------------------------------------------------------------
__global__ __launch_bounds__(32)
void attn_wmma_kernel(const uint16_t* __restrict__ qp,  // (B*QLEN) x HIDDEN bf16
                      const uint16_t* __restrict__ kp,  // (B*KLEN) x HIDDEN bf16
                      const uint16_t* __restrict__ vt,  // HIDDEN x (B*KLEN) bf16 (V^T)
                      uint16_t* __restrict__ op)        // (B*QLEN) x HIDDEN bf16
{
  __shared__ uint16_t plds[64][72];   // 64x64 P tile, padded pitch

  const int lane = threadIdx.x & 31;
  const int lr   = lane & 15;
  const int hi   = lane >> 4;
  const int q0   = blockIdx.x * 64;
  const int h    = blockIdx.y;
  const int b    = blockIdx.z;
  const float scale = 0.125f;         // 1/sqrt(HEAD_DIM)

  // Q fragments for the whole tile (reused every chunk)
  Frag qf[4][2];
#pragma unroll
  for (int mt = 0; mt < 4; ++mt)
#pragma unroll
    for (int ks = 0; ks < 2; ++ks)
      ld_frag(qf[mt][ks], qp + (size_t)(b * QLEN + q0 + mt * 16 + lr) * HIDDEN
                             + h * HEAD_DIM + ks * 32 + hi * 8);

  v8f o[4][4], mold[4], lsum[4];
#pragma unroll
  for (int mt = 0; mt < 4; ++mt) {
#pragma unroll
    for (int r = 0; r < 8; ++r) { mold[mt][r] = -1e30f; lsum[mt][r] = 0.0f; }
#pragma unroll
    for (int nt = 0; nt < 4; ++nt)
#pragma unroll
      for (int r = 0; r < 8; ++r) o[mt][nt][r] = 0.0f;
  }

  for (int kc = 0; kc < KLEN; kc += 64) {
    // ---- S = Q * K^T : B operand columns are key rows (contiguous in d)
    v8f s[4][4];
#pragma unroll
    for (int mt = 0; mt < 4; ++mt)
#pragma unroll
      for (int nt = 0; nt < 4; ++nt)
#pragma unroll
        for (int r = 0; r < 8; ++r) s[mt][nt][r] = 0.0f;

#pragma unroll
    for (int ks = 0; ks < 2; ++ks) {
      Frag kf[4];
#pragma unroll
      for (int nt = 0; nt < 4; ++nt)
        ld_frag(kf[nt], kp + (size_t)(b * KLEN + kc + nt * 16 + lr) * HIDDEN
                           + h * HEAD_DIM + ks * 32 + hi * 8);
#pragma unroll
      for (int mt = 0; mt < 4; ++mt)
#pragma unroll
        for (int nt = 0; nt < 4; ++nt)
          s[mt][nt] = wmma_bf16(qf[mt][ks], kf[nt], s[mt][nt]);
    }

    // ---- online softmax (rows live across 16-lane halves; xor-shuffle reduce)
#pragma unroll
    for (int mt = 0; mt < 4; ++mt) {
#pragma unroll
      for (int nt = 0; nt < 4; ++nt) s[mt][nt] *= scale;

      v8f mnew, alpha;
#pragma unroll
      for (int r = 0; r < 8; ++r) {
        float m = fmaxf(fmaxf(s[mt][0][r], s[mt][1][r]), fmaxf(s[mt][2][r], s[mt][3][r]));
        m = fmaxf(m, __shfl_xor(m, 1));
        m = fmaxf(m, __shfl_xor(m, 2));
        m = fmaxf(m, __shfl_xor(m, 4));
        m = fmaxf(m, __shfl_xor(m, 8));
        mnew[r]  = fmaxf(mold[mt][r], m);
        alpha[r] = __expf(mold[mt][r] - mnew[r]);
      }
#pragma unroll
      for (int nt = 0; nt < 4; ++nt)
#pragma unroll
        for (int r = 0; r < 8; ++r)
          s[mt][nt][r] = __expf(s[mt][nt][r] - mnew[r]);
#pragma unroll
      for (int r = 0; r < 8; ++r) {
        float t = s[mt][0][r] + s[mt][1][r] + s[mt][2][r] + s[mt][3][r];
        t += __shfl_xor(t, 1);
        t += __shfl_xor(t, 2);
        t += __shfl_xor(t, 4);
        t += __shfl_xor(t, 8);
        lsum[mt][r] = lsum[mt][r] * alpha[r] + t;
      }
#pragma unroll
      for (int nt = 0; nt < 4; ++nt) o[mt][nt] *= alpha;
      mold[mt] = mnew;

      // P tile -> LDS as bf16 (C layout -> row-major 64x64)
#pragma unroll
      for (int nt = 0; nt < 4; ++nt)
#pragma unroll
        for (int r = 0; r < 8; ++r)
          plds[mt * 16 + 8 * hi + r][nt * 16 + lr] = f2bf(s[mt][nt][r]);
    }

    // ---- O += P * V  (V^T rows are contiguous along keys)
#pragma unroll
    for (int ks = 0; ks < 2; ++ks) {
      Frag pf[4], vf[4];
#pragma unroll
      for (int mt = 0; mt < 4; ++mt)
        ld_frag(pf[mt], &plds[mt * 16 + lr][ks * 32 + hi * 8]);
#pragma unroll
      for (int nt = 0; nt < 4; ++nt)
        ld_frag(vf[nt], vt + (size_t)(h * HEAD_DIM + nt * 16 + lr) * (BATCH * KLEN)
                           + b * KLEN + kc + ks * 32 + hi * 8);
#pragma unroll
      for (int mt = 0; mt < 4; ++mt)
#pragma unroll
        for (int nt = 0; nt < 4; ++nt)
          o[mt][nt] = wmma_bf16(pf[mt], vf[nt], o[mt][nt]);
    }
  }

  // ---- epilogue: O / l, store bf16 row-major for the final GEMM
#pragma unroll
  for (int mt = 0; mt < 4; ++mt) {
    v8f inv;
#pragma unroll
    for (int r = 0; r < 8; ++r) inv[r] = 1.0f / lsum[mt][r];
#pragma unroll
    for (int nt = 0; nt < 4; ++nt) {
      const int n = h * HEAD_DIM + nt * 16 + lr;
#pragma unroll
      for (int r = 0; r < 8; ++r) {
        const int m = b * QLEN + q0 + mt * 16 + 8 * hi + r;
        op[(size_t)m * HIDDEN + n] = f2bf(o[mt][nt][r] * inv[r]);
      }
    }
  }
}

// ---------------------------------------------------------------------------
extern "C" void kernel_launch(void* const* d_in, const int* in_sizes, int n_in,
                              void* d_out, int out_size, void* d_ws, size_t ws_size,
                              hipStream_t stream) {
  const float* query = (const float*)d_in[0];
  const float* key   = (const float*)d_in[1];
  const float* value = (const float*)d_in[2];
  const float* w_q   = (const float*)d_in[3];
  const float* w_k   = (const float*)d_in[4];
  const float* w_v   = (const float*)d_in[5];
  const float* w_o   = (const float*)d_in[6];
  float* out = (float*)d_out;

  char* ws = (char*)d_ws;
  const size_t MB = (size_t)1 << 20;
  uint16_t* kb  = (uint16_t*)(ws + 0 * MB);    // 16 MB: key   bf16 (8192x1024)
  uint16_t* vb  = (uint16_t*)(ws + 16 * MB);   // 16 MB: value bf16 (8192x1024)
  uint16_t* qb  = (uint16_t*)(ws + 32 * MB);   //  8 MB: query bf16 (4096x1024)
  uint16_t* wqt = (uint16_t*)(ws + 40 * MB);   //  2 MB: w_q^T bf16 (1024x1024)
  uint16_t* wkt = (uint16_t*)(ws + 42 * MB);
  uint16_t* wvt = (uint16_t*)(ws + 44 * MB);
  uint16_t* wot = (uint16_t*)(ws + 46 * MB);
  uint16_t* qp  = (uint16_t*)(ws + 48 * MB);   //  8 MB: Q proj bf16 (4096x1024)
  uint16_t* kp  = (uint16_t*)(ws + 56 * MB);   // 16 MB: K proj bf16 (8192x1024)
  uint16_t* vt  = (uint16_t*)(ws + 72 * MB);   // 16 MB: V^T proj bf16 (1024x8192)
  uint16_t* ao  = (uint16_t*)(ws + 88 * MB);   //  8 MB: attn out bf16 (4096x1024)

  const dim3 b256(256, 1, 1), b32(32, 1, 1);
  const int nq8 = BATCH * QLEN * HIDDEN / 8;   // 524288
  const int nk8 = BATCH * KLEN * HIDDEN / 8;   // 1048576
  const int nw  = HIDDEN * HIDDEN;             // 1M

  // --- prep: activation converts + transposed bf16 weights
  cvt_bf16_kernel<<<dim3(nq8 / 256), b256, 0, stream>>>(query, qb, nq8);
  cvt_bf16_kernel<<<dim3(nk8 / 256), b256, 0, stream>>>(key,   kb, nk8);
  cvt_bf16_kernel<<<dim3(nk8 / 256), b256, 0, stream>>>(value, vb, nk8);
  cvt_tr_bf16_kernel<<<dim3(nw / 256), b256, 0, stream>>>(w_q, wqt, HIDDEN, HIDDEN);
  cvt_tr_bf16_kernel<<<dim3(nw / 256), b256, 0, stream>>>(w_k, wkt, HIDDEN, HIDDEN);
  cvt_tr_bf16_kernel<<<dim3(nw / 256), b256, 0, stream>>>(w_v, wvt, HIDDEN, HIDDEN);
  cvt_tr_bf16_kernel<<<dim3(nw / 256), b256, 0, stream>>>(w_o, wot, HIDDEN, HIDDEN);

  // --- Q/K/V projections (pure bf16 WMMA GEMMs)
  gemm_bf16_kernel<<<dim3(16,  64), b32, 0, stream>>>(qb, HIDDEN, wqt, HIDDEN,
                                                      qp, 1, HIDDEN, HIDDEN);
  gemm_bf16_kernel<<<dim3(16, 128), b32, 0, stream>>>(kb, HIDDEN, wkt, HIDDEN,
                                                      kp, 1, HIDDEN, HIDDEN);
  gemm_bf16_kernel<<<dim3(16, 128), b32, 0, stream>>>(vb, HIDDEN, wvt, HIDDEN,
                                                      vt, 2, BATCH * KLEN, HIDDEN);
  // --- attention: (Q/64, HEADS, B)
  attn_wmma_kernel<<<dim3(QLEN / 64, HEADS, BATCH), b32, 0, stream>>>(qp, kp, vt, ao);
  // --- output projection, f32 result
  gemm_bf16_kernel<<<dim3(16,  64), b32, 0, stream>>>(ao, HIDDEN, wot, HIDDEN,
                                                      out, 0, HIDDEN, HIDDEN);
}
